// PointNetSetAbstraction_3143916061237
// MI455X (gfx1250) — matrix operand order repaired and checked by reading, not compiled, not run on hardware
//
#include <hip/hip_runtime.h>

typedef __attribute__((ext_vector_type(16))) _Float16 v16h;
typedef __attribute__((ext_vector_type(8)))  float    v8f;

#define BB   8
#define NN   8192
#define SS   2048
#define KK   32
#define BSK  (BB * SS * KK)          // 524288
#define R2   0.09f
#define PPT  (NN / 1024)             // 8 points per thread in FPS

// ---------------------------------------------------------------------------
// Farthest point sampling: one block per batch, 1024 threads, dist in regs.
// Writes fps order, new_xyz [B][S][3] (ws) and new_xyz^T (B,3,S) into d_out.
// ---------------------------------------------------------------------------
__global__ __launch_bounds__(1024) void fps_kernel(const float* __restrict__ xyz,
                                                   int* __restrict__ fps_idx,
                                                   float* __restrict__ nxyz,
                                                   float* __restrict__ out_xyz)
{
    const int b   = blockIdx.x;
    const int tid = threadIdx.x;
    const float* xb = xyz + (size_t)b * 3 * NN;

    __shared__ float wval[32];
    __shared__ int   widx[32];
    __shared__ int   far_sh;

    float px[PPT], py[PPT], pz[PPT], dist[PPT];
#pragma unroll
    for (int p = 0; p < PPT; ++p) {
        int j = p * 1024 + tid;
        px[p] = xb[j];
        py[p] = xb[NN + j];
        pz[p] = xb[2 * NN + j];
        dist[p] = 1e10f;
    }

    int far = 0;
    for (int i = 0; i < SS; ++i) {
        // broadcast centroid (uniform address load)
        float cx = xb[far], cy = xb[NN + far], cz = xb[2 * NN + far];
        if (tid == 0) {
            fps_idx[b * SS + i] = far;
            float* nx = nxyz + ((size_t)b * SS + i) * 3;
            nx[0] = cx; nx[1] = cy; nx[2] = cz;
            out_xyz[(size_t)b * 3 * SS + 0 * SS + i] = cx;
            out_xyz[(size_t)b * 3 * SS + 1 * SS + i] = cy;
            out_xyz[(size_t)b * 3 * SS + 2 * SS + i] = cz;
        }
        float bv = -1.0f; int bi = 0;
#pragma unroll
        for (int p = 0; p < PPT; ++p) {
            float dx = px[p] - cx, dy = py[p] - cy, dz = pz[p] - cz;
            float d  = dx * dx + dy * dy + dz * dz;
            float nd = fminf(dist[p], d);
            dist[p]  = nd;
            int j = p * 1024 + tid;
            if (nd > bv) { bv = nd; bi = j; }     // ascending j -> first-max kept
        }
        // wave32 argmax reduction (tie -> smaller index, matches jnp.argmax)
        for (int off = 16; off; off >>= 1) {
            float ov = __shfl_xor(bv, off, 32);
            int   oi = __shfl_xor(bi, off, 32);
            if (ov > bv || (ov == bv && oi < bi)) { bv = ov; bi = oi; }
        }
        int wv = tid >> 5;
        if ((tid & 31) == 0) { wval[wv] = bv; widx[wv] = bi; }
        __syncthreads();
        if (tid < 32) {
            float v = wval[tid]; int ix = widx[tid];
            for (int off = 16; off; off >>= 1) {
                float ov = __shfl_xor(v, off, 32);
                int   oi = __shfl_xor(ix, off, 32);
                if (ov > v || (ov == v && oi < ix)) { v = ov; ix = oi; }
            }
            if (tid == 0) far_sh = ix;
        }
        __syncthreads();
        far = far_sh;
        __syncthreads();
    }
}

// ---------------------------------------------------------------------------
// Ball query: one wave per (b,s); first K indices (ascending) within radius,
// padded with the first hit (reference semantics).
// ---------------------------------------------------------------------------
__global__ __launch_bounds__(256) void ball_query_kernel(const float* __restrict__ xyz,
                                                         const float* __restrict__ nxyz,
                                                         int* __restrict__ idx)
{
    int gtid = blockIdx.x * blockDim.x + threadIdx.x;
    int w    = gtid >> 5;
    int lane = gtid & 31;
    if (w >= BB * SS) return;
    int b = w / SS;
    const float* xb = xyz + (size_t)b * 3 * NN;
    const float* q  = nxyz + (size_t)w * 3;
    float qx = q[0], qy = q[1], qz = q[2];
    int* row = idx + (size_t)w * KK;

    int cnt = 0;
    int first = NN - 1;
    for (int base = 0; base < NN && cnt < KK; base += 32) {
        int j = base + lane;
        float dx = xb[j] - qx, dy = xb[NN + j] - qy, dz = xb[2 * NN + j] - qz;
        bool pred = (dx * dx + dy * dy + dz * dz) <= R2;
        unsigned mask = (unsigned)__ballot(pred);
        if (cnt == 0 && mask) first = base + __builtin_ctz(mask);
        if (pred) {
            int pos = cnt + __popc(mask & ((1u << lane) - 1));
            if (pos < KK) row[pos] = j;
        }
        cnt += __popc(mask);
    }
    int total = cnt < KK ? cnt : KK;
    if (lane >= total) row[lane] = first;   // KK == 32 == wave size
}

// ---------------------------------------------------------------------------
// Build grouped features H0 (f16): [row=(b,s,k)][96] ; ch0..2 rel-xyz,
// ch3..66 point feats, ch67..95 zero pad.
// ---------------------------------------------------------------------------
__global__ void build_h0_kernel(const float* __restrict__ xyz,
                                const float* __restrict__ points,
                                const float* __restrict__ nxyz,
                                const int* __restrict__ idx,
                                _Float16* __restrict__ H0)
{
    int row = blockIdx.x;
    int c   = threadIdx.x;          // 0..95
    int bs  = row >> 5;
    int b   = bs / SS;
    int j   = idx[row];
    float v;
    if (c < 3)       v = xyz[((size_t)b * 3 + c) * NN + j] - nxyz[(size_t)bs * 3 + c];
    else if (c < 67) v = points[((size_t)b * 64 + (c - 3)) * NN + j];
    else             v = 0.f;
    H0[(size_t)row * 96 + c] = (_Float16)v;
}

// ---------------------------------------------------------------------------
// Convert weights f32 -> f16, zero-pad K dimension.
// ---------------------------------------------------------------------------
__global__ void convw_kernel(const float* __restrict__ W, _Float16* __restrict__ Wh,
                             int Co, int Ci, int Kp)
{
    int t = blockIdx.x * blockDim.x + threadIdx.x;
    if (t >= Co * Kp) return;
    int co = t / Kp, c = t - co * Kp;
    Wh[t] = (_Float16)(c < Ci ? W[co * Ci + c] : 0.f);
}

// ---------------------------------------------------------------------------
// WMMA GEMM: Out[r][co] = sum_k A[r][k] * W[co][k] + bias[co]
// One 16x16 output tile per wave via v_wmma_f32_16x16x32_f16.
// A fragment: lane m = row, 16 consecutive K halves (K+16*half .. +15).
// B fragment: lane n = out-channel column, 16 consecutive K halves.
// ---------------------------------------------------------------------------
__global__ __launch_bounds__(256) void gemm_wmma_kernel(const _Float16* __restrict__ A, int Ka,
                                                        const _Float16* __restrict__ W,
                                                        const float* __restrict__ bias,
                                                        int Co, _Float16* __restrict__ Out,
                                                        int numTiles, int coTiles)
{
    int lane = threadIdx.x & 31;
    int wv   = threadIdx.x >> 5;
    int tile = blockIdx.x * 8 + wv;
    if (tile >= numTiles) return;            // wave-uniform: EXEC all 1s for WMMA
    int rowTile = tile / coTiles;
    int coTile  = tile - rowTile * coTiles;
    long r0  = (long)rowTile * 16;
    int  co0 = coTile * 16;
    int  l15 = lane & 15;
    int  hi  = lane >> 4;

    const _Float16* arow = A + (size_t)(r0 + l15) * Ka + (hi ? 16 : 0);
    const _Float16* wrow = W + (size_t)(co0 + l15) * Ka + (hi ? 16 : 0);

    v8f acc = {};
    for (int k0 = 0; k0 < Ka; k0 += 32) {
        v16h a, bm;
#pragma unroll
        for (int i = 0; i < 8; ++i) {
            a[i]     = arow[k0 + i];
            a[8 + i] = arow[k0 + 8 + i];
        }
#pragma unroll
        for (int i = 0; i < 16; ++i) bm[i] = wrow[k0 + i];
        acc = __builtin_amdgcn_wmma_f32_16x16x32_f16(false, a, false, bm,
                                                     (short)0, acc, false, false);
    }
    float bv  = bias[co0 + l15];
    long  rb  = r0 + (hi ? 8 : 0);
    int   col = co0 + l15;
#pragma unroll
    for (int v = 0; v < 8; ++v)
        Out[(size_t)(rb + v) * Co + col] = (_Float16)(acc[v] + bv);
}

// ---------------------------------------------------------------------------
// Per-channel sum / sumsq partials (deterministic, no atomics).
// part[blk][2*C]: [0..C)=sum, [C..2C)=sumsq
// ---------------------------------------------------------------------------
__global__ __launch_bounds__(256) void stats_kernel(const _Float16* __restrict__ H, int C,
                                                    float* __restrict__ part, int rowsPerBlock)
{
    __shared__ float ls[256], ls2[256];
    int tid  = threadIdx.x;
    int c    = tid & (C - 1);
    int rsub = tid / C;
    int nsub = 256 / C;
    long row0 = (long)blockIdx.x * rowsPerBlock;
    float s = 0.f, s2 = 0.f;
    for (int r = rsub; r < rowsPerBlock; r += nsub) {
        float v = (float)H[(size_t)(row0 + r) * C + c];
        s += v; s2 += v * v;
    }
    ls[tid] = s; ls2[tid] = s2;
    __syncthreads();
    if (tid < C) {
        float t1 = ls[tid], t2 = ls2[tid];
        for (int i = 1; i < nsub; ++i) { t1 += ls[tid + i * C]; t2 += ls2[tid + i * C]; }
        part[(size_t)blockIdx.x * 2 * C + c]     = t1;
        part[(size_t)blockIdx.x * 2 * C + C + c] = t2;
    }
}

__global__ void reduce_stats_kernel(const float* __restrict__ part, float* __restrict__ sums,
                                    int C, int nblk)
{
    int t = threadIdx.x;
    if (t < 2 * C) {
        float s = 0.f;
        for (int i = 0; i < nblk; ++i) s += part[(size_t)i * 2 * C + t];
        sums[t] = s;
    }
}

// ---------------------------------------------------------------------------
// BN + ReLU in place (f16).
// ---------------------------------------------------------------------------
__global__ void bnrelu_kernel(_Float16* __restrict__ H, long total, int C,
                              const float* __restrict__ sums,
                              const float* __restrict__ gamma,
                              const float* __restrict__ beta, float invN)
{
    long i      = (long)blockIdx.x * blockDim.x + threadIdx.x;
    long stride = (long)gridDim.x * blockDim.x;
    for (; i < total; i += stride) {
        int c = (int)(i & (C - 1));
        float mean = sums[c] * invN;
        float var  = sums[C + c] * invN - mean * mean;
        float rstd = rsqrtf(var + 1e-5f);
        float v = ((float)H[i] - mean) * rstd * gamma[c] + beta[c];
        H[i] = (_Float16)(v > 0.f ? v : 0.f);
    }
}

// ---------------------------------------------------------------------------
// Final: BN + ReLU + max over K samples; write (B,128,S) f32 output.
// ---------------------------------------------------------------------------
__global__ __launch_bounds__(128) void final_max_kernel(const _Float16* __restrict__ H3,
                                                        const float* __restrict__ sums,
                                                        const float* __restrict__ gamma,
                                                        const float* __restrict__ beta,
                                                        float* __restrict__ out, float invN)
{
    int bs = blockIdx.x;        // 0..B*S-1
    int c  = threadIdx.x;       // 0..127
    float mean = sums[c] * invN;
    float var  = sums[128 + c] * invN - mean * mean;
    float rstd = rsqrtf(var + 1e-5f);
    float g = gamma[c], be = beta[c];
    const _Float16* base = H3 + (size_t)bs * KK * 128 + c;
    float m = 0.f;              // ReLU lower bound
#pragma unroll 4
    for (int k = 0; k < KK; ++k) {
        float v = ((float)base[(size_t)k * 128] - mean) * rstd * g + be;
        v = v > 0.f ? v : 0.f;
        if (v > m) m = v;
    }
    int b = bs / SS, s = bs - b * SS;
    out[(size_t)b * 128 * SS + (size_t)c * SS + s] = m;
}

// ---------------------------------------------------------------------------
extern "C" void kernel_launch(void* const* d_in, const int* in_sizes, int n_in,
                              void* d_out, int out_size, void* d_ws, size_t ws_size,
                              hipStream_t stream)
{
    const float* xyz    = (const float*)d_in[0];
    const float* points = (const float*)d_in[1];
    const float* W0 = (const float*)d_in[2];
    const float* b0 = (const float*)d_in[3];
    const float* g0 = (const float*)d_in[4];
    const float* be0 = (const float*)d_in[5];
    const float* W1 = (const float*)d_in[6];
    const float* b1 = (const float*)d_in[7];
    const float* g1 = (const float*)d_in[8];
    const float* be1 = (const float*)d_in[9];
    const float* W2 = (const float*)d_in[10];
    const float* b2 = (const float*)d_in[11];
    const float* g2 = (const float*)d_in[12];
    const float* be2 = (const float*)d_in[13];
    float* out = (float*)d_out;

    char* ws = (char*)d_ws;
    size_t off = 0;
    auto alloc = [&](size_t bytes) -> void* {
        void* p = ws + off;
        off += (bytes + 255) & ~(size_t)255;
        return p;
    };
    float*    sums0    = (float*)alloc(256 * 4);
    float*    sums1    = (float*)alloc(256 * 4);
    float*    sums2    = (float*)alloc(256 * 4);
    float*    partials = (float*)alloc((size_t)256 * 256 * 4);
    int*      fps_idx  = (int*)alloc((size_t)BB * SS * 4);
    float*    nxyz     = (float*)alloc((size_t)BB * SS * 3 * 4);
    int*      bidx     = (int*)alloc((size_t)BB * SS * KK * 4);
    _Float16* W0h      = (_Float16*)alloc((size_t)64 * 96 * 2);
    _Float16* W1h      = (_Float16*)alloc((size_t)64 * 64 * 2);
    _Float16* W2h      = (_Float16*)alloc((size_t)128 * 64 * 2);
    _Float16* H0       = (_Float16*)alloc((size_t)BSK * 96 * 2);
    _Float16* H1       = (_Float16*)alloc((size_t)BSK * 64 * 2);
    _Float16* H3       = (_Float16*)alloc((size_t)BSK * 128 * 2);
    _Float16* H2       = H0;   // H0 dead after layer-0 GEMM; reuse

    const float invN = 1.0f / (float)BSK;

    convw_kernel<<<(64 * 96 + 255) / 256, 256, 0, stream>>>(W0, W0h, 64, 67, 96);
    convw_kernel<<<(64 * 64 + 255) / 256, 256, 0, stream>>>(W1, W1h, 64, 64, 64);
    convw_kernel<<<(128 * 64 + 255) / 256, 256, 0, stream>>>(W2, W2h, 128, 64, 64);

    fps_kernel<<<BB, 1024, 0, stream>>>(xyz, fps_idx, nxyz, out);
    ball_query_kernel<<<(BB * SS * 32) / 256, 256, 0, stream>>>(xyz, nxyz, bidx);
    build_h0_kernel<<<BSK, 96, 0, stream>>>(xyz, points, nxyz, bidx, H0);

    // Layer 0: 96 -> 64
    {
        int tiles = (BSK / 16) * (64 / 16);   // 131072
        gemm_wmma_kernel<<<tiles / 8, 256, 0, stream>>>(H0, 96, W0h, b0, 64, H1, tiles, 4);
        stats_kernel<<<256, 256, 0, stream>>>(H1, 64, partials, BSK / 256);
        reduce_stats_kernel<<<1, 256, 0, stream>>>(partials, sums0, 64, 256);
        bnrelu_kernel<<<8192, 256, 0, stream>>>(H1, (long)BSK * 64, 64, sums0, g0, be0, invN);
    }
    // Layer 1: 64 -> 64
    {
        int tiles = (BSK / 16) * (64 / 16);
        gemm_wmma_kernel<<<tiles / 8, 256, 0, stream>>>(H1, 64, W1h, b1, 64, H2, tiles, 4);
        stats_kernel<<<256, 256, 0, stream>>>(H2, 64, partials, BSK / 256);
        reduce_stats_kernel<<<1, 256, 0, stream>>>(partials, sums1, 64, 256);
        bnrelu_kernel<<<8192, 256, 0, stream>>>(H2, (long)BSK * 64, 64, sums1, g1, be1, invN);
    }
    // Layer 2: 64 -> 128, fused BN+ReLU+max into output kernel
    {
        int tiles = (BSK / 16) * (128 / 16);  // 262144
        gemm_wmma_kernel<<<tiles / 8, 256, 0, stream>>>(H2, 64, W2h, b2, 128, H3, tiles, 8);
        stats_kernel<<<256, 256, 0, stream>>>(H3, 128, partials, BSK / 256);
        reduce_stats_kernel<<<1, 256, 0, stream>>>(partials, sums2, 128, 256);
        final_max_kernel<<<BB * SS, 128, 0, stream>>>(H3, sums2, g2, be2,
                                                      out + (size_t)BB * 3 * SS, invN);
    }
}
